// MultiheadAttention_73933567033450
// MI455X (gfx1250) — matrix operand order repaired and checked by reading, not compiled, not run on hardware
//
#include <hip/hip_runtime.h>

// ---------------- CDNA5 WMMA types ----------------
typedef __attribute__((ext_vector_type(16))) __bf16 v16bf;
typedef __attribute__((ext_vector_type(8)))  float  v8f;

union Frag16 {            // 16 bf16 = 32 bytes = one WMMA A/B operand (wave32)
  v16bf bf;
  uint4 u4[2];
};

static __device__ __forceinline__ v8f wmma_bf16(const Frag16& a, const Frag16& b, v8f c) {
  return __builtin_amdgcn_wmma_f32_16x16x32_bf16(false, a.bf, false, b.bf,
                                                 (short)0, c, false, false);
}

// fp32 -> bf16 with round-to-nearest-even
static __device__ __forceinline__ unsigned short f2bf(float f) {
  unsigned u = __float_as_uint(f);
  u = (u + 0x7FFFu + ((u >> 16) & 1u)) >> 16;
  return (unsigned short)u;
}
static __device__ __forceinline__ unsigned pack_bf16(float a, float b) {
  return (unsigned)f2bf(a) | ((unsigned)f2bf(b) << 16);
}

// ---------------- problem constants ----------------
constexpr int Bn = 4, T = 2048, Cc = 768, H = 12, Dd = 64;
constexpr int Mrows = Bn * T;                 // 8192
constexpr int BM = 128, BN = 128, BK = 32;
constexpr int NSTEP = Cc / BK;                // 24 k-steps

// =====================================================================
// Kernel 1: fused QKV projection.  y = x @ W + b, cast to bf16.
//   z==0 -> Q*scale [B,H,T,D]  z==1 -> K [B,H,T,D]  z==2 -> V^T [B,H,D,T]
// Double-buffered LDS; global loads for tile k+1 issued during compute k;
// B-fragment LDS loads software-pipelined past each WMMA.
// =====================================================================
__global__ __launch_bounds__(256) void qkv_kernel(
    const float* __restrict__ x,
    const float* __restrict__ wq, const float* __restrict__ bq,
    const float* __restrict__ wk, const float* __restrict__ bk,
    const float* __restrict__ wv, const float* __restrict__ bvv,
    unsigned short* __restrict__ Qb, unsigned short* __restrict__ Kb,
    unsigned short* __restrict__ Vt)
{
  __shared__ unsigned int As[2][BM * 16];   // ping-pong: 128 rows x 32 bf16
  __shared__ unsigned int Bt[2][BN * 16];   // ping-pong: 128 cols x 32 bf16 (W^T)

  const int which = blockIdx.z;
  const float* w    = (which == 0) ? wq : ((which == 1) ? wk : wv);
  const float* bias = (which == 0) ? bq : ((which == 1) ? bk : bvv);

  const int m0 = blockIdx.x * BM;
  const int n0 = blockIdx.y * BN;

  const int wave = threadIdx.x >> 5;
  const int lane = threadIdx.x & 31;
  const int lm   = lane & 15;
  const int lh   = lane >> 4;

  v8f acc[8];
#pragma unroll
  for (int nt = 0; nt < 8; ++nt) { v8f z = {0.f,0.f,0.f,0.f,0.f,0.f,0.f,0.f}; acc[nt] = z; }

  const int r    = threadIdx.x >> 1;   // staging: row of A tile
  const int half = threadIdx.x & 1;    // staging: which 16-wide K slice
  const int nsc  = threadIdx.x & 127;  // staging: W column
  const int kq   = threadIdx.x >> 7;   // staging: 0/1 -> k sub-block of 16

  const float* xrow = x + (size_t)(m0 + r) * Cc + half * 16;
  const float* wcol = w + (size_t)(kq * 16) * Cc + n0 + nsc;

  float4 fA0, fA1, fA2, fA3;
  float  fW[16];

  auto load_regs = [&](int kt) {
    const float4* src = (const float4*)(xrow + kt * BK);
    fA0 = src[0]; fA1 = src[1]; fA2 = src[2]; fA3 = src[3];
    const float* wp = wcol + (size_t)kt * BK * Cc;
#pragma unroll
    for (int i = 0; i < 16; ++i) fW[i] = wp[i * Cc];
  };
  auto store_tile = [&](int b) {
    uint4 p0 = make_uint4(pack_bf16(fA0.x, fA0.y), pack_bf16(fA0.z, fA0.w),
                          pack_bf16(fA1.x, fA1.y), pack_bf16(fA1.z, fA1.w));
    uint4 p1 = make_uint4(pack_bf16(fA2.x, fA2.y), pack_bf16(fA2.z, fA2.w),
                          pack_bf16(fA3.x, fA3.y), pack_bf16(fA3.z, fA3.w));
    uint4* dst = (uint4*)&As[b][r * 16 + half * 8];
    dst[0] = p0; dst[1] = p1;
#pragma unroll
    for (int i = 0; i < 8; ++i)
      Bt[b][nsc * 16 + kq * 8 + i] = pack_bf16(fW[2 * i], fW[2 * i + 1]);
  };

  load_regs(0);
  store_tile(0);
  __syncthreads();
  load_regs(1);
  int buf = 0;

  for (int kt = 0; kt < NSTEP; ++kt) {
    // --- compute from buffer `buf`; pipeline B-frag loads past WMMAs ---
    Frag16 fa, fb[2];
    const uint4* pa = (const uint4*)&As[buf][(wave * 16 + lm) * 16];
    fa.u4[0] = pa[2 * lh];      // K = 16h..16h+7
    fa.u4[1] = pa[2 * lh + 1];  // K = 16h+8..16h+15
    {
      const uint4* pb = (const uint4*)&Bt[buf][lm * 16];
      fb[0].u4[0] = pb[2 * lh];
      fb[0].u4[1] = pb[2 * lh + 1];
    }
#pragma unroll
    for (int nt = 0; nt < 8; ++nt) {
      if (nt + 1 < 8) {   // issue next fragment's loads before this WMMA
        const uint4* pb = (const uint4*)&Bt[buf][((nt + 1) * 16 + lm) * 16];
        fb[(nt + 1) & 1].u4[0] = pb[2 * lh];
        fb[(nt + 1) & 1].u4[1] = pb[2 * lh + 1];
      }
      acc[nt] = wmma_bf16(fa, fb[nt & 1], acc[nt]);
    }
    // --- stage tile kt+1 into the other buffer, issue loads for kt+2 ---
    if (kt + 1 < NSTEP) {
      store_tile(buf ^ 1);
      __syncthreads();
      if (kt + 2 < NSTEP) load_regs(kt + 2);
      buf ^= 1;
    }
  }

  // --- epilogue: bias add, (Q pre-scaled by D^-1/2), bf16 scatter ---
  const float oscale = (which == 0) ? 0.125f : 1.0f;
  const int m_base = m0 + wave * 16 + lh * 8;
#pragma unroll
  for (int nt = 0; nt < 8; ++nt) {
    const int n  = n0 + nt * 16 + lm;
    const float bval = bias[n];
    const int hh = n >> 6;          // head
    const int d  = n & 63;          // dim within head
#pragma unroll
    for (int i = 0; i < 8; ++i) {
      const int m = m_base + i;
      const int b = m >> 11;        // T = 2048
      const int t = m & 2047;
      const unsigned short o = f2bf((acc[nt][i] + bval) * oscale);
      if (which == 2)
        Vt[((size_t)(b * H + hh) * Dd + d) * T + t] = o;
      else if (which == 1)
        Kb[((size_t)(b * H + hh) * T + t) * Dd + d] = o;
      else
        Qb[((size_t)(b * H + hh) * T + t) * Dd + d] = o;
    }
  }
}

// =====================================================================
// Kernel 2: flash attention.  4 waves/block, 16 queries per wave.
//   S = Qs K^T (scale folded into Q) ; online softmax ; O = P V
// =====================================================================
__global__ __launch_bounds__(128) void attn_kernel(
    const unsigned short* __restrict__ Qb, const unsigned short* __restrict__ Kb,
    const unsigned short* __restrict__ Vt, unsigned short* __restrict__ Ab)
{
  __shared__ unsigned short Ps[4 * 16 * 64];   // per-wave P transpose scratch (8KB)

  const int wave = threadIdx.x >> 5;
  const int lane = threadIdx.x & 31;
  const int lm   = lane & 15;
  const int lh   = lane >> 4;

  const int bh = blockIdx.y;                   // fused (batch, head)
  const int q0 = blockIdx.x * 64 + wave * 16;

  const unsigned short* Qh = Qb + (size_t)bh * T * Dd;
  const unsigned short* Kh = Kb + (size_t)bh * T * Dd;
  const unsigned short* Vh = Vt + (size_t)bh * Dd * T;
  unsigned short* myP = &Ps[wave * 16 * 64];

  // Q fragments (2 k-steps of 32 over D=64), loaded once
  Frag16 fq[2];
  {
    const uint4* p = (const uint4*)(Qh + (size_t)(q0 + lm) * Dd);
    fq[0].u4[0] = p[2 * lh];     fq[0].u4[1] = p[2 * lh + 1];
    fq[1].u4[0] = p[4 + 2 * lh]; fq[1].u4[1] = p[5 + 2 * lh];
  }

  float mi[8], li[8];
  v8f accO[4];
#pragma unroll
  for (int i = 0; i < 8; ++i) { mi[i] = -1e30f; li[i] = 0.f; }
#pragma unroll
  for (int nt = 0; nt < 4; ++nt) { v8f z = {0.f,0.f,0.f,0.f,0.f,0.f,0.f,0.f}; accO[nt] = z; }

  for (int kb = 0; kb < T / 64; ++kb) {
    const int kt0 = kb * 64;

    // prefetch next key-block (each K/V row = one 128B cacheline)
    if (kb + 1 < T / 64) {
      const int nx = kt0 + 64;
      __builtin_prefetch(Kh + (size_t)(nx + lane) * Dd, 0, 3);
      __builtin_prefetch(Kh + (size_t)(nx + 32 + lane) * Dd, 0, 3);
      __builtin_prefetch(Vh + (size_t)lane * T + nx, 0, 3);
      __builtin_prefetch(Vh + (size_t)(32 + lane) * T + nx, 0, 3);
    }

    // ---- S = Qs K^T over this 64-key block ----
    v8f s[4];
#pragma unroll
    for (int nt = 0; nt < 4; ++nt) { v8f z = {0.f,0.f,0.f,0.f,0.f,0.f,0.f,0.f}; s[nt] = z; }
#pragma unroll
    for (int nt = 0; nt < 4; ++nt) {
      const uint4* pk4 = (const uint4*)(Kh + (size_t)(kt0 + nt * 16 + lm) * Dd);
#pragma unroll
      for (int kd = 0; kd < 2; ++kd) {
        Frag16 fk;
        fk.u4[0] = pk4[kd * 4 + 2 * lh];
        fk.u4[1] = pk4[kd * 4 + 2 * lh + 1];
        s[nt] = wmma_bf16(fq[kd], fk, s[nt]);
      }
    }

    // ---- online softmax: row max / rescale / row sum ----
    float alpha[8];
#pragma unroll
    for (int i = 0; i < 8; ++i) {
      float v = fmaxf(fmaxf(s[0][i], s[1][i]), fmaxf(s[2][i], s[3][i]));
#pragma unroll
      for (int off = 1; off < 16; off <<= 1) v = fmaxf(v, __shfl_xor(v, off, 32));
      const float mnew = fmaxf(mi[i], v);
      alpha[i] = __expf(mi[i] - mnew);
      mi[i] = mnew;
    }
    float rs[8];
#pragma unroll
    for (int i = 0; i < 8; ++i) rs[i] = 0.f;
#pragma unroll
    for (int nt = 0; nt < 4; ++nt)
#pragma unroll
      for (int i = 0; i < 8; ++i) {
        const float p = __expf(s[nt][i] - mi[i]);
        s[nt][i] = p;
        rs[i] += p;
      }
#pragma unroll
    for (int i = 0; i < 8; ++i) {
      float v = rs[i];
#pragma unroll
      for (int off = 1; off < 16; off <<= 1) v += __shfl_xor(v, off, 32);
      li[i] = li[i] * alpha[i] + v;
    }
#pragma unroll
    for (int nt = 0; nt < 4; ++nt)
#pragma unroll
      for (int i = 0; i < 8; ++i) accO[nt][i] *= alpha[i];

    // ---- transpose P (C-frag layout) -> A-frag layout via LDS ----
#pragma unroll
    for (int nt = 0; nt < 4; ++nt)
#pragma unroll
      for (int i = 0; i < 8; ++i)
        myP[(8 * lh + i) * 64 + nt * 16 + lm] = f2bf(s[nt][i]);

    Frag16 fp[2];
    {
      const uint4* pp = (const uint4*)(myP + lm * 64);
      fp[0].u4[0] = pp[2 * lh];     fp[0].u4[1] = pp[2 * lh + 1];
      fp[1].u4[0] = pp[4 + 2 * lh]; fp[1].u4[1] = pp[5 + 2 * lh];
    }

    // ---- O += P @ V  (V stored [D,T] so token-pairs are contiguous) ----
#pragma unroll
    for (int nt = 0; nt < 4; ++nt) {
      const uint4* pv = (const uint4*)(Vh + (size_t)(nt * 16 + lm) * T + kt0);
#pragma unroll
      for (int kd = 0; kd < 2; ++kd) {
        Frag16 fv;
        fv.u4[0] = pv[kd * 4 + 2 * lh];
        fv.u4[1] = pv[kd * 4 + 2 * lh + 1];
        accO[nt] = wmma_bf16(fp[kd], fv, accO[nt]);
      }
    }
  }

  // ---- normalize and write bf16 [B,T,C] ----
  const int b  = bh / H;
  const int hh = bh % H;
#pragma unroll
  for (int i = 0; i < 8; ++i) {
    const float inv = 1.0f / li[i];
    const int row = q0 + 8 * lh + i;
#pragma unroll
    for (int nt = 0; nt < 4; ++nt)
      Ab[((size_t)(b * T + row)) * Cc + hh * 64 + nt * 16 + lm] = f2bf(accO[nt][i] * inv);
  }
}

// =====================================================================
// Kernel 3: output projection.  out = attn(bf16) @ wo + bo   (fp32 out)
// =====================================================================
__global__ __launch_bounds__(256) void proj_kernel(
    const unsigned short* __restrict__ Ab,
    const float* __restrict__ wo, const float* __restrict__ bo,
    float* __restrict__ out)
{
  __shared__ unsigned int As[2][BM * 16];
  __shared__ unsigned int Bt[2][BN * 16];

  const int m0 = blockIdx.x * BM;
  const int n0 = blockIdx.y * BN;

  const int wave = threadIdx.x >> 5;
  const int lane = threadIdx.x & 31;
  const int lm   = lane & 15;
  const int lh   = lane >> 4;

  v8f acc[8];
#pragma unroll
  for (int nt = 0; nt < 8; ++nt) { v8f z = {0.f,0.f,0.f,0.f,0.f,0.f,0.f,0.f}; acc[nt] = z; }

  const int r    = threadIdx.x >> 1;
  const int half = threadIdx.x & 1;
  const int nsc  = threadIdx.x & 127;
  const int kq   = threadIdx.x >> 7;

  const unsigned short* arow = Ab + (size_t)(m0 + r) * Cc + half * 16;
  const float* wcol = wo + (size_t)(kq * 16) * Cc + n0 + nsc;

  uint4 rA0, rA1;
  float fW[16];

  auto load_regs = [&](int kt) {
    const uint4* src = (const uint4*)(arow + kt * BK);
    rA0 = src[0]; rA1 = src[1];
    const float* wp = wcol + (size_t)kt * BK * Cc;
#pragma unroll
    for (int i = 0; i < 16; ++i) fW[i] = wp[i * Cc];
  };
  auto store_tile = [&](int b) {
    uint4* dst = (uint4*)&As[b][r * 16 + half * 8];
    dst[0] = rA0; dst[1] = rA1;
#pragma unroll
    for (int i = 0; i < 8; ++i)
      Bt[b][nsc * 16 + kq * 8 + i] = pack_bf16(fW[2 * i], fW[2 * i + 1]);
  };

  load_regs(0);
  store_tile(0);
  __syncthreads();
  load_regs(1);
  int buf = 0;

  for (int kt = 0; kt < NSTEP; ++kt) {
    Frag16 fa, fb[2];
    const uint4* pa = (const uint4*)&As[buf][(wave * 16 + lm) * 16];
    fa.u4[0] = pa[2 * lh];
    fa.u4[1] = pa[2 * lh + 1];
    {
      const uint4* pb = (const uint4*)&Bt[buf][lm * 16];
      fb[0].u4[0] = pb[2 * lh];
      fb[0].u4[1] = pb[2 * lh + 1];
    }
#pragma unroll
    for (int nt = 0; nt < 8; ++nt) {
      if (nt + 1 < 8) {   // issue next fragment's loads before this WMMA
        const uint4* pb = (const uint4*)&Bt[buf][((nt + 1) * 16 + lm) * 16];
        fb[(nt + 1) & 1].u4[0] = pb[2 * lh];
        fb[(nt + 1) & 1].u4[1] = pb[2 * lh + 1];
      }
      acc[nt] = wmma_bf16(fa, fb[nt & 1], acc[nt]);
    }
    if (kt + 1 < NSTEP) {
      store_tile(buf ^ 1);
      __syncthreads();
      if (kt + 2 < NSTEP) load_regs(kt + 2);
      buf ^= 1;
    }
  }

  const int m_base = m0 + wave * 16 + lh * 8;
#pragma unroll
  for (int nt = 0; nt < 8; ++nt) {
    const int n = n0 + nt * 16 + lm;
    const float bval = bo[n];
#pragma unroll
    for (int i = 0; i < 8; ++i)
      out[(size_t)(m_base + i) * Cc + n] = acc[nt][i] + bval;
  }
}

// =====================================================================
extern "C" void kernel_launch(void* const* d_in, const int* in_sizes, int n_in,
                              void* d_out, int out_size, void* d_ws, size_t ws_size,
                              hipStream_t stream) {
  const float* x   = (const float*)d_in[0];
  const float* wq  = (const float*)d_in[1];
  const float* bq  = (const float*)d_in[2];
  const float* wk  = (const float*)d_in[3];
  const float* bk  = (const float*)d_in[4];
  const float* wv  = (const float*)d_in[5];
  const float* bvv = (const float*)d_in[6];
  const float* wo  = (const float*)d_in[7];
  const float* bo  = (const float*)d_in[8];
  float* out = (float*)d_out;

  const size_t nqkv = (size_t)Bn * H * T * Dd;          // 6,291,456 elements
  unsigned short* Qb = (unsigned short*)d_ws;           // [B,H,T,D] bf16 (pre-scaled)
  unsigned short* Kb = Qb + nqkv;                       // [B,H,T,D] bf16
  unsigned short* Vt = Kb + nqkv;                       // [B,H,D,T] bf16
  unsigned short* Ab = Vt + nqkv;                       // [B,T,C]   bf16

  qkv_kernel<<<dim3(Mrows / BM, Cc / BN, 3), 256, 0, stream>>>(
      x, wq, bq, wk, bk, wv, bvv, Qb, Kb, Vt);
  attn_kernel<<<dim3(T / 64, Bn * H), 128, 0, stream>>>(Qb, Kb, Vt, Ab);
  proj_kernel<<<dim3(Mrows / BM, Cc / BN), 256, 0, stream>>>(Ab, wo, bo, out);
}
